// SimpleGNN_33956011442634
// MI455X (gfx1250) — compile-verified, matrix operand order checked by reading, not run on hardware
//
#include <hip/hip_runtime.h>

#define N_NODES 100000
#define N_EDGES 1600000
#define BN_EPS 1e-5f

typedef __attribute__((ext_vector_type(16))) _Float16 v16h;
typedef __attribute__((ext_vector_type(8)))  _Float16 v8h;
typedef __attribute__((ext_vector_type(8)))  float    v8f;

// ---------- float max via monotonic unsigned encoding ----------
__device__ __forceinline__ unsigned encf(float f) {
    unsigned u = __float_as_uint(f);
    return (u & 0x80000000u) ? ~u : (u | 0x80000000u);
}
__device__ __forceinline__ float decf(unsigned u) {
    return __uint_as_float((u & 0x80000000u) ? (u ^ 0x80000000u) : ~u);
}

// ---------- small init kernels ----------
__global__ void init_small_kernel(float* stats, float* z, float* pooled, unsigned* gmaxkey) {
    int t = threadIdx.x;
    for (int i = t; i < 384; i += blockDim.x) stats[i] = 0.f;
    for (int i = t; i < 64; i += blockDim.x) { z[i] = 0.f; gmaxkey[i] = 0x007FFFFFu; } // enc(-inf)
    for (int i = t; i < 4096; i += blockDim.x) pooled[i] = 0.f;
}

__global__ void deg_init_kernel(float* dinv) {
    int i = blockIdx.x * blockDim.x + threadIdx.x;
    if (i < N_NODES) dinv[i] = 1.0f;   // self loop
}

__global__ void deg_edge_kernel(const int* __restrict__ ei, float* dinv) {
    int e = blockIdx.x * blockDim.x + threadIdx.x;
    if (e < N_EDGES) atomicAdd(&dinv[ei[N_EDGES + e]], 1.0f);
}

__global__ void deg_fin_kernel(float* dinv) {
    int i = blockIdx.x * blockDim.x + threadIdx.x;
    if (i < N_NODES) dinv[i] = rsqrtf(dinv[i]);
}

// ---------- one-time weight convert: f32 [k][n] -> f16 [n][k] ----------
__global__ void convert_w_kernel(const float* __restrict__ W1, const float* __restrict__ W2,
                                 _Float16* __restrict__ w1t, _Float16* __restrict__ w2t) {
    int t = blockIdx.x * blockDim.x + threadIdx.x;
    if (t < 256 * 128) {
        int k = t / 128, n = t % 128;
        w1t[n * 256 + k] = (_Float16)W1[t];
    }
    if (t < 128 * 64) {
        int k = t / 64, n = t % 64;
        w2t[n * 128 + k] = (_Float16)W2[t];
    }
}

// ---------- WMMA GEMM: Y[M x NO] = X[M x KTOT] * W[KTOT x NO], f16 in / f32 acc ----------
// Wt is pre-transposed f16 [NO][KTOT]. One block = 8 waves; wave w owns M-tile
// (blockIdx*8+w) and all NO/16 N-tiles. Wt chunks are staged to LDS with
// GLOBAL_LOAD_ASYNC_TO_LDS_B128 (ASYNCcnt), so B fragments are two contiguous
// ds_load_b128 per lane matching the 16-bit WMMA B layout.
template <int KTOT, int NO, int KSTAGE>
__global__ void gemm_wmma_kernel(const float* __restrict__ X, const _Float16* __restrict__ Wt,
                                 float* __restrict__ Y, int M) {
    __shared__ __align__(16) _Float16 sW[NO * KSTAGE]; // [n][k]
    const int tid  = threadIdx.x;
    const int wave = tid >> 5;
    const int lane = tid & 31;
    const int l15  = lane & 15;
    const int mtile = blockIdx.x * 8 + wave;
    const bool valid = (mtile * 16 < M);
    const int m0 = (valid ? mtile : 0) * 16;
    const int ksel = (lane < 16) ? 0 : 8;

    constexpr int NT = NO / 16;
    v8f acc[NT];
#pragma unroll
    for (int n = 0; n < NT; ++n)
#pragma unroll
        for (int r = 0; r < 8; ++r) acc[n][r] = 0.f;

    const int row = m0 + l15;

    for (int kh = 0; kh < KTOT / KSTAGE; ++kh) {
        // ---- async stage of the f16 weight chunk into LDS (16B granules) ----
        constexpr int CHUNKS = NO * KSTAGE / 8;
        for (int c = tid; c < CHUNKS; c += 256) {
            int n  = c / (KSTAGE / 8);
            int kc = c % (KSTAGE / 8);
            const _Float16* g = Wt + (size_t)n * KTOT + kh * KSTAGE + kc * 8;
            unsigned lofs = (unsigned)(size_t)&sW[n * KSTAGE + kc * 8];
            asm volatile("global_load_async_to_lds_b128 %0, %1, off"
                         :: "v"(lofs), "v"(g) : "memory");
        }
        asm volatile("s_wait_asynccnt 0" ::: "memory");
        __syncthreads();

        const float* xrow = X + (size_t)row * KTOT + kh * KSTAGE;
#pragma unroll
        for (int kk = 0; kk < KSTAGE / 32; ++kk) {
            // A fragment: 16-bit A 16x32 layout (lanes 0-15: K 0-7/16-23; lanes 16-31: +8)
            const float* ap = xrow + kk * 32 + ksel;
            v16h a;
#pragma unroll
            for (int j = 0; j < 8; ++j) a[j] = (_Float16)ap[j];
#pragma unroll
            for (int j = 0; j < 8; ++j) a[8 + j] = (_Float16)ap[16 + j];
#pragma unroll
            for (int n = 0; n < NT; ++n) {
                const _Float16* bp = &sW[(n * 16 + l15) * KSTAGE + kk * 32 + ksel];
                v8h blo = *(const v8h*)(bp);
                v8h bhi = *(const v8h*)(bp + 16);
                v16h b;
#pragma unroll
                for (int j = 0; j < 8; ++j) { b[j] = blo[j]; b[8 + j] = bhi[j]; }
                acc[n] = __builtin_amdgcn_wmma_f32_16x16x32_f16(
                    false, a, false, b, (short)0, acc[n], false, false);
            }
        }
        __syncthreads();
    }

    if (valid) {
        const int mrow = m0 + ((lane < 16) ? 0 : 8);
#pragma unroll
        for (int n = 0; n < NT; ++n)
#pragma unroll
            for (int r = 0; r < 8; ++r)
                Y[(size_t)(mrow + r) * NO + n * 16 + l15] = acc[n][r];
    }
}

// ---------- aggregation: A[c] = sum_{e: col==c} H[row]*dinv[row]*dinv[col] + H[c]*dinv[c]^2 + bias ----------
template <int F>
__global__ void agg_init_kernel(const float* __restrict__ H, const float* __restrict__ dinv,
                                const float* __restrict__ bias, float* __restrict__ A) {
    size_t t = (size_t)blockIdx.x * blockDim.x + threadIdx.x;
    if (t >= (size_t)N_NODES * F) return;
    int i = (int)(t / F), f = (int)(t % F);
    float d = dinv[i];
    A[t] = H[t] * d * d + bias[f];
}

template <int F>
__global__ void agg_edge_kernel(const float* __restrict__ H, const float* __restrict__ dinv,
                                const int* __restrict__ ei, float* __restrict__ A) {
    constexpr int LPE = F / 4;
    size_t t = (size_t)blockIdx.x * blockDim.x + threadIdx.x;
    if (t >= (size_t)N_EDGES * LPE) return;
    int e  = (int)(t / LPE);
    int f0 = (int)(t % LPE) * 4;
    int r = ei[e], c = ei[N_EDGES + e];
    float norm = dinv[r] * dinv[c];
    float4 h = *(const float4*)(H + (size_t)r * F + f0);
    float* out = A + (size_t)c * F + f0;
    atomicAdd(out + 0, h.x * norm);
    atomicAdd(out + 1, h.y * norm);
    atomicAdd(out + 2, h.z * norm);
    atomicAdd(out + 3, h.w * norm);
}

// ---------- batchnorm over axis 0 ----------
template <int F>
__global__ void bn_stats_kernel(const float* __restrict__ A, float* sum, float* sq) {
    __shared__ float ssum[256], ssq[256];
    int tid = threadIdx.x;
    int c = tid % F;
    constexpr int RPB = 256 / F;
    float s = 0.f, q = 0.f;
    for (int r = blockIdx.x * RPB + tid / F; r < N_NODES; r += gridDim.x * RPB) {
        float v = A[(size_t)r * F + c];
        s += v; q += v * v;
    }
    ssum[tid] = s; ssq[tid] = q;
    __syncthreads();
    if (tid < F) {
        for (int j = F; j < 256; j += F) { s += ssum[tid + j]; q += ssq[tid + j]; }
        atomicAdd(&sum[c], s);
        atomicAdd(&sq[c], q);
    }
}

template <int F>
__global__ void bn_apply_kernel(float* __restrict__ A, const float* sum, const float* sq,
                                const float* gamma, const float* beta) {
    size_t t = (size_t)blockIdx.x * blockDim.x + threadIdx.x;
    if (t >= (size_t)N_NODES * F) return;
    int f = (int)(t % F);
    float mu  = sum[f] * (1.0f / N_NODES);
    float var = sq[f] * (1.0f / N_NODES) - mu * mu;
    float v = (A[t] - mu) * rsqrtf(var + BN_EPS) * gamma[f] + beta[f];
    A[t] = v > 0.f ? v : 0.f;
}

// ---------- gate MLP + per-graph max ----------
__global__ void gate_kernel(const float* __restrict__ X3, const float* Wg1, const float* bg1,
                            const float* Wg2, const float* bg2, const int* __restrict__ batch,
                            float* __restrict__ gate, unsigned* gmaxkey) {
    __shared__ float sW1[64 * 32], sb1[32], sW2[32], sb2;
    int tid = threadIdx.x;
    for (int i = tid; i < 64 * 32; i += blockDim.x) sW1[i] = Wg1[i];
    if (tid < 32) { sb1[tid] = bg1[tid]; sW2[tid] = Wg2[tid]; }
    if (tid == 0) sb2 = bg2[0];
    __syncthreads();
    int i = blockIdx.x * blockDim.x + tid;
    if (i >= N_NODES) return;
    const float* x = X3 + (size_t)i * 64;
    float xr[64];
#pragma unroll
    for (int k = 0; k < 64; ++k) xr[k] = x[k];
    float g = sb2;
    for (int j = 0; j < 32; ++j) {
        float h = sb1[j];
#pragma unroll
        for (int k = 0; k < 64; ++k) h += xr[k] * sW1[k * 32 + j];
        h = h > 0.f ? h : 0.f;
        g += h * sW2[j];
    }
    gate[i] = g;
    atomicMax(&gmaxkey[batch[i]], encf(g));
}

__global__ void expz_kernel(float* __restrict__ gate, const int* __restrict__ batch,
                            const unsigned* gmaxkey, float* z) {
    int i = blockIdx.x * blockDim.x + threadIdx.x;
    if (i >= N_NODES) return;
    int b = batch[i];
    float e = expf(gate[i] - decf(gmaxkey[b]));
    gate[i] = e;
    atomicAdd(&z[b], e);
}

__global__ void pool_kernel(const float* __restrict__ X3, const float* __restrict__ gate,
                            const int* __restrict__ batch, const float* z, float* pooled) {
    size_t t = (size_t)blockIdx.x * blockDim.x + threadIdx.x;
    if (t >= (size_t)N_NODES * 16) return;
    int i = (int)(t >> 4);
    int f0 = (int)(t & 15) * 4;
    int b = batch[i];
    float w = gate[i] / z[b];
    float4 h = *(const float4*)(X3 + (size_t)i * 64 + f0);
    float* p = pooled + b * 64 + f0;
    atomicAdd(p + 0, h.x * w);
    atomicAdd(p + 1, h.y * w);
    atomicAdd(p + 2, h.z * w);
    atomicAdd(p + 3, h.w * w);
}

__global__ void head_kernel(const float* pooled, const float* Wc1, const float* bc1,
                            const float* Wc2, const float* bc2, float* out) {
    int g = threadIdx.x;
    if (g >= 64) return;
    const float* p = pooled + g * 64;
    float h[32];
    for (int j = 0; j < 32; ++j) {
        float s = bc1[j];
        for (int k = 0; k < 64; ++k) s += p[k] * Wc1[k * 32 + j];
        h[j] = s > 0.f ? s : 0.f;
    }
    for (int c = 0; c < 2; ++c) {
        float s = bc2[c];
        for (int j = 0; j < 32; ++j) s += h[j] * Wc2[j * 2 + c];
        out[g * 2 + c] = s;
    }
}

extern "C" void kernel_launch(void* const* d_in, const int* in_sizes, int n_in,
                              void* d_out, int out_size, void* d_ws, size_t ws_size,
                              hipStream_t stream) {
    const float* x      = (const float*)d_in[0];
    const int*   ei     = (const int*)d_in[1];
    const int*   batch  = (const int*)d_in[2];
    const float* W1     = (const float*)d_in[3];
    const float* b1     = (const float*)d_in[4];
    const float* gamma1 = (const float*)d_in[5];
    const float* beta1  = (const float*)d_in[6];
    const float* W2     = (const float*)d_in[7];
    const float* b2     = (const float*)d_in[8];
    const float* gamma2 = (const float*)d_in[9];
    const float* beta2  = (const float*)d_in[10];
    const float* Wg1    = (const float*)d_in[11];
    const float* bg1    = (const float*)d_in[12];
    const float* Wg2    = (const float*)d_in[13];
    const float* bg2    = (const float*)d_in[14];
    const float* Wc1    = (const float*)d_in[15];
    const float* bc1    = (const float*)d_in[16];
    const float* Wc2    = (const float*)d_in[17];
    const float* bc2    = (const float*)d_in[18];

    float* f_H   = (float*)d_ws;                        // N*128 (H1; later H2 + X3)
    float* f_A   = f_H + (size_t)N_NODES * 128;         // N*128 (A1 -> X2)
    float* dinv  = f_A + (size_t)N_NODES * 128;         // N
    float* gate  = dinv + N_NODES;                      // N
    float* stats = gate + N_NODES;                      // 384
    float* zbuf  = stats + 384;                         // 64
    float* pooled = zbuf + 64;                          // 4096
    unsigned* gmaxkey = (unsigned*)(pooled + 4096);     // 64
    _Float16* w1t = (_Float16*)(gmaxkey + 64);          // 256*128 halfs (f16 W1^T [n][k])
    _Float16* w2t = w1t + 256 * 128;                    // 128*64  halfs (f16 W2^T [n][k])
    float* H2 = f_H;                                    // N*64
    float* X3 = f_H + (size_t)N_NODES * 64;             // N*64 (A2 -> X3)

    const int TB = 256;
    auto blks = [](size_t n, int b) { return (int)((n + b - 1) / b); };

    init_small_kernel<<<1, 512, 0, stream>>>(stats, zbuf, pooled, gmaxkey);
    convert_w_kernel<<<128, TB, 0, stream>>>(W1, W2, w1t, w2t);
    deg_init_kernel<<<blks(N_NODES, TB), TB, 0, stream>>>(dinv);
    deg_edge_kernel<<<blks(N_EDGES, TB), TB, 0, stream>>>(ei, dinv);
    deg_fin_kernel<<<blks(N_NODES, TB), TB, 0, stream>>>(dinv);

    // Layer 1
    gemm_wmma_kernel<256, 128, 128><<<782, TB, 0, stream>>>(x, w1t, f_H, N_NODES);
    agg_init_kernel<128><<<blks((size_t)N_NODES * 128, TB), TB, 0, stream>>>(f_H, dinv, b1, f_A);
    agg_edge_kernel<128><<<blks((size_t)N_EDGES * 32, TB), TB, 0, stream>>>(f_H, dinv, ei, f_A);
    bn_stats_kernel<128><<<256, TB, 0, stream>>>(f_A, stats + 0, stats + 128);
    bn_apply_kernel<128><<<blks((size_t)N_NODES * 128, TB), TB, 0, stream>>>(f_A, stats + 0, stats + 128, gamma1, beta1);

    // Layer 2
    gemm_wmma_kernel<128, 64, 128><<<782, TB, 0, stream>>>(f_A, w2t, H2, N_NODES);
    agg_init_kernel<64><<<blks((size_t)N_NODES * 64, TB), TB, 0, stream>>>(H2, dinv, b2, X3);
    agg_edge_kernel<64><<<blks((size_t)N_EDGES * 16, TB), TB, 0, stream>>>(H2, dinv, ei, X3);
    bn_stats_kernel<64><<<256, TB, 0, stream>>>(X3, stats + 256, stats + 320);
    bn_apply_kernel<64><<<blks((size_t)N_NODES * 64, TB), TB, 0, stream>>>(X3, stats + 256, stats + 320, gamma2, beta2);

    // Attention pooling + head
    gate_kernel<<<blks(N_NODES, TB), TB, 0, stream>>>(X3, Wg1, bg1, Wg2, bg2, batch, gate, gmaxkey);
    expz_kernel<<<blks(N_NODES, TB), TB, 0, stream>>>(gate, batch, gmaxkey, zbuf);
    pool_kernel<<<blks((size_t)N_NODES * 16, TB), TB, 0, stream>>>(X3, gate, batch, zbuf, pooled);
    head_kernel<<<1, 64, 0, stream>>>(pooled, Wc1, bc1, Wc2, bc2, (float*)d_out);
}
